// BasicSelfAttention_86577950752777
// MI455X (gfx1250) — matrix operand order repaired and checked by reading, not compiled, not run on hardware
//
#include <hip/hip_runtime.h>
#include <hip/hip_bf16.h>

// ---------------------------------------------------------------------------
// Flash-attention for B=8, S=4096, D=64 (fp32 in/out) on gfx1250 (MI455X).
//  - Q*K^T and P*V via v_wmma_f32_16x16x32_f16 (f16 in, f32 accumulate)
//  - Softmax row-sums on the matrix pipe (extra WMMA with B = ones).
//  - Packed-f16 row-max xor-reduction (half the ds_bpermute traffic).
//  - Global->LDS tile loads software-pipelined (register double buffer).
//  - LDS rows padded to 72 halves (144B): with 64 banks x 4B, row*36 mod 64
//    sweeps all multiples of 4 -> staging stores & fragment loads are spread
//    across banks instead of aliasing to 4-8 banks.
// ---------------------------------------------------------------------------

typedef _Float16 v16h __attribute__((ext_vector_type(16)));
typedef _Float16 v8h  __attribute__((ext_vector_type(8)));
typedef _Float16 v4h  __attribute__((ext_vector_type(4)));
typedef _Float16 h2   __attribute__((ext_vector_type(2)));
typedef float    v8f  __attribute__((ext_vector_type(8)));

#define S_LEN 4096
#define DHEAD 64
#define TK    64          // key tile (keys per LDS stage)
#define WAVES 8           // waves per block; each wave owns 16 query rows
#define LPAD  72          // padded LDS row stride in halves (144B)

__device__ __forceinline__ v8f wmma_f16(v16h a, v16h b, v8f c) {
  return __builtin_amdgcn_wmma_f32_16x16x32_f16(
      /*neg_a=*/false, a, /*neg_b=*/false, b,
      /*c_mod=*/(short)0, c, /*reuse_a=*/false, /*reuse_b=*/false);
}

// 16 contiguous halves from LDS as two 16B ds_load_b128 (16B-aligned).
__device__ __forceinline__ v16h lds_frag16(const _Float16* p) {
  const v8h lo = *(const v8h*)p;
  const v8h hi = *(const v8h*)(p + 8);
  return __builtin_shufflevector(lo, hi, 0, 1, 2, 3, 4, 5, 6, 7,
                                 8, 9, 10, 11, 12, 13, 14, 15);
}

__global__ __launch_bounds__(256) void
fa_fwd_kernel(const float* __restrict__ X, float* __restrict__ Y) {
  // LDS: 9KB + 9KB + 18KB = 36KB per workgroup (320KB WGP pool).
  __shared__ __align__(16) _Float16 kv [TK * LPAD];        // [key][dim]
  __shared__ __align__(16) _Float16 kvT[DHEAD * LPAD];     // [dim][key]
  __shared__ __align__(16) _Float16 ptile[WAVES][16 * LPAD]; // P [row][key]

  const int tid  = threadIdx.x;
  const int wave = tid >> 5;
  const int lane = tid & 31;
  const int hi   = lane >> 4;   // half-wave select (0/1)
  const int ln   = lane & 15;

  const int b     = blockIdx.x >> 5;          // 32 blocks per batch
  const int qblk  = blockIdx.x & 31;
  const int qbase = qblk * (WAVES * 16) + wave * 16;

  const float* Xb = X + (size_t)b * S_LEN * DHEAD;

  // ---- Load Q A-fragments (16x64 rows qbase..qbase+15), f32 -> f16 ----
  // A layout (16-bit, 16x32): lane holds row M=ln; elems 0..15 -> K=hi*16+0..15
  // (contiguous, matching B).  Two K-chunks of 32 cover D=64.
  v16h qa[2];
  {
    const float* qrow = Xb + (size_t)(qbase + ln) * DHEAD;
#pragma unroll
    for (int kc = 0; kc < 2; ++kc) {
      const int base = kc * 32 + hi * 16;
#pragma unroll
      for (int j = 0; j < 8; ++j) {
        qa[kc][j]     = (_Float16)qrow[base + j];
        qa[kc][8 + j] = (_Float16)qrow[base + 8 + j];
      }
    }
  }

  // All-ones f16 B operand: one extra WMMA per K-chunk accumulates the
  // softmax row-sums into lacc (replicated across the 16 columns).
  v16h ones;
#pragma unroll
  for (int j = 0; j < 16; ++j) ones[j] = (_Float16)1.0f;

  // ---- Accumulators: O (16x64) as four 16x16 f32 tiles; l as one tile. ----
  v8f o[4], lacc;
  float m[8];
  {
    v8f z = {0.f, 0.f, 0.f, 0.f, 0.f, 0.f, 0.f, 0.f};
#pragma unroll
    for (int c = 0; c < 4; ++c) o[c] = z;
    lacc = z;
  }
#pragma unroll
  for (int r = 0; r < 8; ++r) m[r] = -1.0e30f;

  // Cooperative tile-load slice: thread t owns key = t&63, dims base+(0..15).
  // Per store instruction the dim is lane-uniform and keys are consecutive,
  // so kvT column stores hit 16 distinct banks (2B merges within words).
  const int ld_key  = tid & 63;
  const int ld_dim0 = (tid >> 6) * 16;

  const int NT = S_LEN / TK;

  // Prologue of the software pipeline: fetch tile 0 into registers.
  float4 stage[4];
#pragma unroll
  for (int j = 0; j < 4; ++j)
    stage[j] = *(const float4*)(Xb + (size_t)ld_key * DHEAD + ld_dim0 + j * 4);

  for (int kt = 0; kt < NT; ++kt) {
    // ---- Drain register stage into LDS (f32 -> f16, both layouts). ----
#pragma unroll
    for (int j = 0; j < 4; ++j) {
      const float4 v = stage[j];
      const int dim = ld_dim0 + j * 4;
      const _Float16 h0 = (_Float16)v.x, h1 = (_Float16)v.y;
      const _Float16 h2v = (_Float16)v.z, h3 = (_Float16)v.w;
      v4h pk = {h0, h1, h2v, h3};
      *(v4h*)&kv[ld_key * LPAD + dim] = pk;        // row-major (for Q*K^T)
      kvT[(dim + 0) * LPAD + ld_key] = h0;         // transposed (for P*V)
      kvT[(dim + 1) * LPAD + ld_key] = h1;
      kvT[(dim + 2) * LPAD + ld_key] = h2v;
      kvT[(dim + 3) * LPAD + ld_key] = h3;
    }
    __syncthreads();

    // Issue next tile's HBM loads now; latency hides under this tile's math.
    if (kt + 1 < NT) {
#pragma unroll
      for (int j = 0; j < 4; ++j)
        stage[j] = *(const float4*)(Xb + (size_t)((kt + 1) * TK + ld_key) * DHEAD + ld_dim0 + j * 4);
    }

    // ---- Scores: S(16x64) = Q(16x64) @ K_tile^T, 4 N-chunks x 2 K-chunks --
    // B layout (16-bit, 32x16): lane gives N=ln (key), elems 0..15 give
    // K = hi*16 + 0..15 (dims), contiguous in the kv row.
    v8f s[4];
#pragma unroll
    for (int n = 0; n < 4; ++n) {
      v8f z = {0.f, 0.f, 0.f, 0.f, 0.f, 0.f, 0.f, 0.f};
      s[n] = z;
#pragma unroll
      for (int kc = 0; kc < 2; ++kc) {
        const v16h bf = lds_frag16(&kv[(n * 16 + ln) * LPAD + kc * 32 + hi * 16]);
        s[n] = wmma_f16(qa[kc], bf, s[n]);
      }
    }

    // ---- Row max (C/D layout: vgpr r -> row r + hi*8, lane -> col). ----
    // Packed f16x2 xor-reduction across the 16-lane column groups; any
    // lane-uniform near-max is algebraically exact for online softmax.
    float tmax[8];
#pragma unroll
    for (int r = 0; r < 8; ++r) {
      float t = s[0][r];
#pragma unroll
      for (int n = 1; n < 4; ++n) t = fmaxf(t, s[n][r]);
      tmax[r] = t;
    }
    h2 pk2[4];
#pragma unroll
    for (int j = 0; j < 4; ++j) {
      h2 p = {(_Float16)tmax[2 * j], (_Float16)tmax[2 * j + 1]};
      pk2[j] = p;
    }
#pragma unroll
    for (int mask = 1; mask <= 8; mask <<= 1)
#pragma unroll
      for (int j = 0; j < 4; ++j) {
        const int other = __shfl_xor(__builtin_bit_cast(int, pk2[j]), mask, 32);
        pk2[j] = __builtin_elementwise_max(pk2[j], __builtin_bit_cast(h2, other));
      }

    float corr[8];
#pragma unroll
    for (int r = 0; r < 8; ++r) {
      const float mx = (float)pk2[r >> 1][r & 1];
      const float mn = fmaxf(m[r], mx);
      corr[r] = __expf(m[r] - mn);
      m[r]    = mn;
    }

    // ---- P = exp(S - m) -> per-wave LDS scratch in [row][key] order. ----
    _Float16* pw = &ptile[wave][0];
#pragma unroll
    for (int n = 0; n < 4; ++n)
#pragma unroll
      for (int r = 0; r < 8; ++r) {
        const float pv = __expf(s[n][r] - m[r]);
        pw[(r + hi * 8) * LPAD + n * 16 + ln] = (_Float16)pv;
      }

    // Rescale running accumulators (O and l share the same layout).
#pragma unroll
    for (int c = 0; c < 4; ++c)
#pragma unroll
      for (int r = 0; r < 8; ++r) o[c][r] *= corr[r];
#pragma unroll
    for (int r = 0; r < 8; ++r) lacc[r] *= corr[r];

    // Wave-private LDS RAW: wait for our own ds stores (lockstep wave32).
    asm volatile("s_wait_dscnt 0x0" ::: "memory");

    // ---- Re-load P as A-operand fragments (two K-chunks of 32 keys). ----
    v16h pa[2];
#pragma unroll
    for (int kc = 0; kc < 2; ++kc) {
      const _Float16* pr = &pw[ln * LPAD + kc * 32 + hi * 16];
      const v8h lo  = *(const v8h*)pr;
      const v8h hi8 = *(const v8h*)(pr + 8);
      pa[kc] = __builtin_shufflevector(lo, hi8, 0, 1, 2, 3, 4, 5, 6, 7,
                                       8, 9, 10, 11, 12, 13, 14, 15);
    }

    // ---- l += rowsum(P) on the matrix pipe (B = ones). ----
    lacc = wmma_f16(pa[0], ones, lacc);
    lacc = wmma_f16(pa[1], ones, lacc);

    // ---- O += P(16x64) @ V(64x64): B from kvT (N=dims, K=keys). ----
#pragma unroll
    for (int c = 0; c < 4; ++c)
#pragma unroll
      for (int kc = 0; kc < 2; ++kc) {
        const v16h bf = lds_frag16(&kvT[(c * 16 + ln) * LPAD + kc * 32 + hi * 16]);
        o[c] = wmma_f16(pa[kc], bf, o[c]);
      }

    __syncthreads();  // protect kv/kvT before next tile overwrites
  }

  // ---- Epilogue: Y = O / l  (l replicated across columns in lacc). ----
  float rl[8];
#pragma unroll
  for (int r = 0; r < 8; ++r) rl[r] = 1.0f / lacc[r];
#pragma unroll
  for (int c = 0; c < 4; ++c)
#pragma unroll
    for (int r = 0; r < 8; ++r) {
      const size_t row = (size_t)b * S_LEN + qbase + r + hi * 8;
      Y[row * DHEAD + c * 16 + ln] = o[c][r] * rl[r];
    }
}

extern "C" void kernel_launch(void* const* d_in, const int* in_sizes, int n_in,
                              void* d_out, int out_size, void* d_ws, size_t ws_size,
                              hipStream_t stream) {
  (void)in_sizes; (void)n_in; (void)out_size; (void)d_ws; (void)ws_size;
  const float* X = (const float*)d_in[0];
  float* Y = (float*)d_out;
  // 8 batches * (4096 / 128 rows-per-block) = 256 blocks of 256 threads.
  dim3 grid(256), block(256);
  fa_fwd_kernel<<<grid, block, 0, stream>>>(X, Y);
}
